// CausalAttention_70171175682131
// MI455X (gfx1250) — matrix operand order repaired and checked by reading, not compiled, not run on hardware
//
#include <hip/hip_runtime.h>
#include <hip/hip_bf16.h>

// ---------------------------------------------------------------------------
// Causal multi-head attention block (GPT-2 style) for gfx1250 / MI455X.
// bf16 WMMA (16x16x32, f32 accum) everywhere; flash attention with
// double-buffered async global->LDS K/V staging (ASYNCcnt pipeline).
// ---------------------------------------------------------------------------

#define N_EMBD 768
#define N_HEAD 12
#define HS     64
#define TSEQ   1024
#define BATCH  8

typedef __attribute__((ext_vector_type(16))) __bf16 v16bf;
typedef __attribute__((ext_vector_type(8)))  __bf16 v8bf;
typedef __attribute__((ext_vector_type(8)))  float  v8f;
typedef __attribute__((__vector_size__(16))) int    vi4;   // 4 x i32

#define AS1 __attribute__((address_space(1)))
#define AS3 __attribute__((address_space(3)))

#if defined(__has_builtin)
#if __has_builtin(__builtin_amdgcn_global_load_async_to_lds_b128)
#define HAVE_ASYNC 1
#endif
#endif
#ifndef HAVE_ASYNC
#define HAVE_ASYNC 0
#endif

#if defined(__has_builtin) && __has_builtin(__builtin_amdgcn_s_wait_asynccnt)
#define WAIT_ASYNC(n) __builtin_amdgcn_s_wait_asynccnt(n)
#else
#define WAIT_ASYNC(n) asm volatile("s_wait_asynccnt " #n ::: "memory")
#endif

#if HAVE_ASYNC
// One 16-byte per-lane async copy global -> LDS (tracked on ASYNCcnt).
// Builtin signature (from hipcc diagnostic): (v4i AS1*, v4i AS3*, Ii, Ii).
__device__ __forceinline__ void cp_async16(const unsigned short* g,
                                           unsigned short* l) {
  __builtin_amdgcn_global_load_async_to_lds_b128((AS1 vi4*)g, (AS3 vi4*)l,
                                                 0, 0);
}
#endif

__device__ __forceinline__ unsigned short f2bf(float x) {
  union { float f; unsigned int u; } c; c.f = x;
  unsigned int r = (c.u + 0x7fffu + ((c.u >> 16) & 1u)) >> 16;  // RNE
  return (unsigned short)r;
}

// Build a v16bf A/B fragment from two 16-byte (8-half) chunks.
__device__ __forceinline__ v16bf ld_frag(const unsigned short* p0,
                                         const unsigned short* p1) {
  union { v16bf v; v8bf h[2]; } u;
  u.h[0] = *(const v8bf*)p0;
  u.h[1] = *(const v8bf*)p1;
  return u.v;
}

// ---------------------------------------------------------------------------
// Shared WMMA GEMM mainloop: block tile 128(M) x 128(N), K-step 32.
// 256 threads = 8 waves arranged 4(M) x 2(N); each wave owns a 32x64 tile
// (2x4 fragments of 16x16) -> 8 WMMAs per wave per K-step.
//   sA[row][k]  row-major, stride 40 halves (16B-aligned chunks)
//   sB[col][k]  (W transposed), 16 contiguous K-halves per lane.
// Staging is split load-phase / store-phase so global loads issue
// back-to-back and drain with descending s_wait_loadcnt.
// ---------------------------------------------------------------------------
template <bool ABF16>
__device__ __forceinline__ void gemm_main(const void* __restrict__ Xv, int ldx,
                                          const float* __restrict__ W, int ldw,
                                          int m0, int n0,
                                          unsigned short* sA,
                                          unsigned short* sB,
                                          v8f acc[2][4]) {
  const int tid  = threadIdx.x;
  const int lane = tid & 31, wid = tid >> 5;
  const int lo = lane & 15, hi = lane >> 4;
  const int wm = (wid >> 1) * 32, wn = (wid & 1) * 64;

#pragma unroll
  for (int fm = 0; fm < 2; ++fm)
#pragma unroll
    for (int fn = 0; fn < 4; ++fn)
#pragma unroll
      for (int i = 0; i < 8; ++i) acc[fm][fn][i] = 0.0f;

  for (int k0 = 0; k0 < N_EMBD; k0 += 32) {
    __syncthreads();
    // ---- Stage A tile: 128 rows x 32 K (bf16) ----
    if (ABF16) {
#if HAVE_ASYNC
      // A already bf16 row-major: async-copy 512 x 16B chunks to LDS.
#pragma unroll
      for (int i = 0; i < 2; ++i) {
        int idx = tid + i * 256;
        int row = idx >> 2, seg = idx & 3;
        cp_async16((const unsigned short*)Xv + (size_t)(m0 + row) * ldx + k0 +
                       seg * 8,
                   sA + row * 40 + seg * 8);
      }
#else
      uint2 a[4];
#pragma unroll
      for (int i = 0; i < 4; ++i) {
        int idx = tid + i * 256;
        int row = idx >> 3, c4 = idx & 7;
        a[i] = *(const uint2*)((const unsigned short*)Xv +
                               (size_t)(m0 + row) * ldx + k0 + c4 * 4);
      }
#pragma unroll
      for (int i = 0; i < 4; ++i) {
        int idx = tid + i * 256;
        int row = idx >> 3, c4 = idx & 7;
        unsigned int* dst = (unsigned int*)(sA + row * 40 + c4 * 4);
        dst[0] = a[i].x; dst[1] = a[i].y;
      }
#endif
    } else {
      float4 f[4];
#pragma unroll
      for (int i = 0; i < 4; ++i) {
        int idx = tid + i * 256;
        int row = idx >> 3, c4 = idx & 7;
        f[i] = *(const float4*)((const float*)Xv +
                                (size_t)(m0 + row) * ldx + k0 + c4 * 4);
      }
#pragma unroll
      for (int i = 0; i < 4; ++i) {
        int idx = tid + i * 256;
        int row = idx >> 3, c4 = idx & 7;
        unsigned int* dst = (unsigned int*)(sA + row * 40 + c4 * 4);
        dst[0] = (unsigned)f2bf(f[i].x) | ((unsigned)f2bf(f[i].y) << 16);
        dst[1] = (unsigned)f2bf(f[i].z) | ((unsigned)f2bf(f[i].w) << 16);
      }
    }
    // ---- Stage B tile transposed: sB[n][k] <- W[k0+k][n0+n] ----
    float w[16];
#pragma unroll
    for (int i = 0; i < 16; ++i) {
      int idx = tid + i * 256;
      int nn = idx & 127, kk = idx >> 7;
      w[i] = W[(size_t)(k0 + kk) * ldw + (n0 + nn)];
    }
#pragma unroll
    for (int i = 0; i < 16; ++i) {
      int idx = tid + i * 256;
      int nn = idx & 127, kk = idx >> 7;
      sB[nn * 40 + kk] = f2bf(w[i]);
    }
    // Prefetch next K-slice of W into caches while we compute.
    if (k0 + 32 < N_EMBD)
      __builtin_prefetch(&W[(size_t)(k0 + 32) * ldw + n0 + (tid & 127)], 0, 1);
#if HAVE_ASYNC
    if (ABF16) WAIT_ASYNC(0);
#endif
    __syncthreads();

    v16bf af[2], bfr[4];
#pragma unroll
    for (int fm = 0; fm < 2; ++fm) {
      // A 16x32 layout: lanes 0-15 -> K {0..15}; 16-31 -> {16..31}
      const unsigned short* base = sA + (wm + fm * 16 + lo) * 40;
      af[fm] = ld_frag(base + hi * 16, base + hi * 16 + 8);
    }
#pragma unroll
    for (int fn = 0; fn < 4; ++fn) {
      // B 32x16 layout: lane = column, 16 contiguous K-halves.
      const unsigned short* base = sB + (wn + fn * 16 + lo) * 40 + hi * 16;
      bfr[fn] = ld_frag(base, base + 8);
    }
#pragma unroll
    for (int fm = 0; fm < 2; ++fm)
#pragma unroll
      for (int fn = 0; fn < 4; ++fn)
        acc[fm][fn] = __builtin_amdgcn_wmma_f32_16x16x32_bf16(
            false, af[fm], false, bfr[fn], (short)0, acc[fm][fn], false, false);
  }
}

// ---------------------------------------------------------------------------
// Kernel 1: QKV = X @ W_attn + b_attn -> bf16 Q[B,H,T,hs], K[B,H,T,hs],
// V transposed [B,H,hs,T].
// ---------------------------------------------------------------------------
__global__ __launch_bounds__(256) void qkv_gemm_kernel(
    const float* __restrict__ X, const float* __restrict__ Wa,
    const float* __restrict__ ba, unsigned short* __restrict__ Qb,
    unsigned short* __restrict__ Kb, unsigned short* __restrict__ Vt) {
  __shared__ __align__(16) unsigned short sA[128 * 40];
  __shared__ __align__(16) unsigned short sB[128 * 40];
  const int m0 = blockIdx.x * 128, n0 = blockIdx.y * 128;
  v8f acc[2][4];
  gemm_main<false>(X, N_EMBD, Wa, 3 * N_EMBD, m0, n0, sA, sB, acc);

  const int lane = threadIdx.x & 31, wid = threadIdx.x >> 5;
  const int lo = lane & 15, hi = lane >> 4;
  const int wm = (wid >> 1) * 32, wn = (wid & 1) * 64;
#pragma unroll
  for (int fm = 0; fm < 2; ++fm)
#pragma unroll
    for (int fn = 0; fn < 4; ++fn)
#pragma unroll
      for (int r = 0; r < 8; ++r) {
        int gm = m0 + wm + fm * 16 + r + 8 * hi;   // row in [0, B*T)
        int gn = n0 + wn + fn * 16 + lo;           // col in [0, 3C)
        float v = acc[fm][fn][r] + ba[gn];
        unsigned short bv = f2bf(v);
        int b = gm >> 10, t = gm & 1023;
        int part = gn / N_EMBD, cn = gn % N_EMBD;
        int h = cn >> 6, d = cn & 63;
        size_t bh = (size_t)b * N_HEAD + h;
        if (part == 0)      Qb[(bh * TSEQ + t) * HS + d] = bv;
        else if (part == 1) Kb[(bh * TSEQ + t) * HS + d] = bv;
        else                Vt[(bh * HS + d) * TSEQ + t] = bv;
      }
}

// ---------------------------------------------------------------------------
// Kernel 2: flash attention.  One wave per (b,h, 16-query tile); 32-key
// blocks, K/V tiles double-buffered in LDS via async global->LDS copies.
// ---------------------------------------------------------------------------
__global__ __launch_bounds__(128) void flash_attn_kernel(
    const unsigned short* __restrict__ Qb, const unsigned short* __restrict__ Kb,
    const unsigned short* __restrict__ Vt, unsigned short* __restrict__ Ybf) {
  __shared__ __align__(16) unsigned short pbuf[4][16][32];
#if HAVE_ASYNC
  __shared__ __align__(16) unsigned short kbuf[4][2][32 * 64]; // [wave][buf][key][hs]
  __shared__ __align__(16) unsigned short vbuf[4][2][64 * 32]; // [wave][buf][hs][key]
#endif
  const int tid = threadIdx.x;
  const int lane = tid & 31, wid = tid >> 5;
  const int lo = lane & 15, hi = lane >> 4;
  const int tile = blockIdx.x * 4 + wid;   // 6144 tiles total
  const int bh = tile >> 6;
  const int q0 = (tile & 63) * 16;
  const unsigned short* Q = Qb + (size_t)bh * TSEQ * HS;
  const unsigned short* K = Kb + (size_t)bh * TSEQ * HS;
  const unsigned short* V = Vt + (size_t)bh * HS * TSEQ;
  unsigned short(*pb)[32] = pbuf[wid];

  // Q fragments (both hs chunks), live across the whole key loop.
  v16bf qf[2];
#pragma unroll
  for (int c = 0; c < 2; ++c) {
    const unsigned short* base = Q + (size_t)(q0 + lo) * HS + c * 32;
    qf[c] = ld_frag(base + hi * 16, base + hi * 16 + 8);
  }

  v8f O[4];
  float mrow[8], lrow[8];
#pragma unroll
  for (int c = 0; c < 4; ++c)
#pragma unroll
    for (int i = 0; i < 8; ++i) O[c][i] = 0.0f;
#pragma unroll
  for (int r = 0; r < 8; ++r) { mrow[r] = -1e30f; lrow[r] = 0.0f; }

  const float scale = 0.125f;  // 1/sqrt(64)

#if HAVE_ASYNC
  unsigned short* kbw = kbuf[wid][0];
  unsigned short* vbw = vbuf[wid][0];
  // Issue async copies of one 32-key K tile (4KB) + V tile (4KB) into buf.
  auto issue = [&](int buf, int j0) {
    unsigned short* kd = kbw + buf * 2048;
    const unsigned short* kg = K + (size_t)j0 * HS;      // contiguous 4KB
#pragma unroll
    for (int i = 0; i < 8; ++i)
      cp_async16(kg + lane * 8 + i * 256, kd + lane * 8 + i * 256);
    unsigned short* vd = vbw + buf * 2048;
#pragma unroll
    for (int i = 0; i < 8; ++i) {
      int c = lane + 32 * i;            // 256 chunks of 16B
      int r = c >> 2, seg = c & 3;      // hs row, 16B segment of 32 keys
      cp_async16(V + (size_t)r * TSEQ + j0 + seg * 8, vd + r * 32 + seg * 8);
    }
  };
  issue(0, 0);
#endif

  for (int j0 = 0; j0 <= q0 + 15; j0 += 32) {
#if HAVE_ASYNC
    const int buf = (j0 >> 5) & 1;
    const bool more = (j0 + 32) <= (q0 + 15);      // wave-uniform
    if (more) { issue(buf ^ 1, j0 + 32); WAIT_ASYNC(16); }
    else      { WAIT_ASYNC(0); }
    const unsigned short* kb = kbw + buf * 2048;   // [key][hs]
    const unsigned short* vb = vbw + buf * 2048;   // [hs][key]
#endif
    // K fragments: [key-half][hs-chunk], B-layout (lane = key column).
    v16bf kf[2][2];
#pragma unroll
    for (int kh = 0; kh < 2; ++kh)
#pragma unroll
      for (int c = 0; c < 2; ++c) {
#if HAVE_ASYNC
        const unsigned short* base =
            kb + (size_t)(kh * 16 + lo) * HS + c * 32 + hi * 16;
#else
        const unsigned short* base =
            K + (size_t)(j0 + kh * 16 + lo) * HS + c * 32 + hi * 16;
#endif
        kf[kh][c] = ld_frag(base, base + 8);
      }
    v8f S0, S1;
#pragma unroll
    for (int i = 0; i < 8; ++i) { S0[i] = 0.0f; S1[i] = 0.0f; }
    S0 = __builtin_amdgcn_wmma_f32_16x16x32_bf16(false, qf[0], false, kf[0][0], (short)0, S0, false, false);
    S0 = __builtin_amdgcn_wmma_f32_16x16x32_bf16(false, qf[1], false, kf[0][1], (short)0, S0, false, false);
    S1 = __builtin_amdgcn_wmma_f32_16x16x32_bf16(false, qf[0], false, kf[1][0], (short)0, S1, false, false);
    S1 = __builtin_amdgcn_wmma_f32_16x16x32_bf16(false, qf[1], false, kf[1][1], (short)0, S1, false, false);

    // Online softmax per C-fragment row (row r+8*hi spans 16 lanes).
#pragma unroll
    for (int r = 0; r < 8; ++r) {
      const int qi = q0 + r + 8 * hi;
      float s0 = S0[r] * scale;
      float s1 = S1[r] * scale;
      if (j0 + lo > qi)      s0 = -1e30f;   // causal mask
      if (j0 + 16 + lo > qi) s1 = -1e30f;
      float mx = fmaxf(s0, s1);
#pragma unroll
      for (int d = 1; d < 16; d <<= 1) mx = fmaxf(mx, __shfl_xor(mx, d, 32));
      float mnew  = fmaxf(mrow[r], mx);
      float alpha = __expf(mrow[r] - mnew);
      mrow[r] = mnew;
      float p0 = __expf(s0 - mnew);
      float p1 = __expf(s1 - mnew);
      float rs = p0 + p1;
#pragma unroll
      for (int d = 1; d < 16; d <<= 1) rs += __shfl_xor(rs, d, 32);
      lrow[r] = lrow[r] * alpha + rs;
#pragma unroll
      for (int c = 0; c < 4; ++c) O[c][r] *= alpha;
      pb[r + 8 * hi][lo]      = f2bf(p0);
      pb[r + 8 * hi][16 + lo] = f2bf(p1);
    }

    // P fragment (A 16x32 layout) from per-wave LDS scratch.
    v16bf pf = ld_frag(&pb[lo][hi * 16], &pb[lo][hi * 16 + 8]);
#pragma unroll
    for (int c = 0; c < 4; ++c) {
#if HAVE_ASYNC
      const unsigned short* vp = vb + (size_t)(c * 16 + lo) * 32 + hi * 16;
#else
      const unsigned short* vp = V + (size_t)(c * 16 + lo) * TSEQ + j0 + hi * 16;
#endif
      v16bf vf = ld_frag(vp, vp + 8);
      O[c] = __builtin_amdgcn_wmma_f32_16x16x32_bf16(false, pf, false, vf, (short)0, O[c], false, false);
    }
  }

  const int b = bh / N_HEAD, h = bh % N_HEAD;
#pragma unroll
  for (int c = 0; c < 4; ++c)
#pragma unroll
    for (int r = 0; r < 8; ++r) {
      float val = O[c][r] / lrow[r];
      int t = q0 + r + 8 * hi;
      Ybf[((size_t)b * TSEQ + t) * N_EMBD + h * HS + c * 16 + lo] = f2bf(val);
    }
}

// ---------------------------------------------------------------------------
// Kernel 3: out = Y @ W_proj + b_proj  (bf16 A input, f32 output).
// ---------------------------------------------------------------------------
__global__ __launch_bounds__(256) void proj_gemm_kernel(
    const unsigned short* __restrict__ Ybf, const float* __restrict__ Wp,
    const float* __restrict__ bp, float* __restrict__ Out) {
  __shared__ __align__(16) unsigned short sA[128 * 40];
  __shared__ __align__(16) unsigned short sB[128 * 40];
  const int m0 = blockIdx.x * 128, n0 = blockIdx.y * 128;
  v8f acc[2][4];
  gemm_main<true>(Ybf, N_EMBD, Wp, N_EMBD, m0, n0, sA, sB, acc);

  const int lane = threadIdx.x & 31, wid = threadIdx.x >> 5;
  const int lo = lane & 15, hi = lane >> 4;
  const int wm = (wid >> 1) * 32, wn = (wid & 1) * 64;
#pragma unroll
  for (int fm = 0; fm < 2; ++fm)
#pragma unroll
    for (int fn = 0; fn < 4; ++fn)
#pragma unroll
      for (int r = 0; r < 8; ++r) {
        int gm = m0 + wm + fm * 16 + r + 8 * hi;
        int gn = n0 + wn + fn * 16 + lo;
        Out[(size_t)gm * N_EMBD + gn] = acc[fm][fn][r] + bp[gn];
      }
}

// ---------------------------------------------------------------------------
extern "C" void kernel_launch(void* const* d_in, const int* in_sizes, int n_in,
                              void* d_out, int out_size, void* d_ws,
                              size_t ws_size, hipStream_t stream) {
  const float* x      = (const float*)d_in[0];
  const float* w_attn = (const float*)d_in[1];
  const float* b_attn = (const float*)d_in[2];
  const float* w_proj = (const float*)d_in[3];
  const float* b_proj = (const float*)d_in[4];
  float* out = (float*)d_out;

  char* ws = (char*)d_ws;
  const size_t SZ = (size_t)BATCH * N_HEAD * TSEQ * HS * sizeof(unsigned short);
  unsigned short* Qb  = (unsigned short*)(ws);
  unsigned short* Kb  = (unsigned short*)(ws + SZ);
  unsigned short* Vt  = (unsigned short*)(ws + 2 * SZ);
  unsigned short* Ybf = (unsigned short*)(ws + 3 * SZ);
  (void)in_sizes; (void)n_in; (void)out_size; (void)ws_size;

  // QKV GEMM: M=8192 (64 tiles), N=2304 (18 tiles of 128)
  qkv_gemm_kernel<<<dim3(64, 18), dim3(256), 0, stream>>>(x, w_attn, b_attn,
                                                          Qb, Kb, Vt);
  // Flash attention: 8*12*64 = 6144 wave-tiles / 4 waves per block
  flash_attn_kernel<<<dim3(1536), dim3(128), 0, stream>>>(Qb, Kb, Vt, Ybf);
  // Projection GEMM: M=8192 (64 tiles), N=768 (6 tiles of 128)
  proj_gemm_kernel<<<dim3(64, 6), dim3(256), 0, stream>>>(Ybf, w_proj, b_proj,
                                                          out);
}
